// HSTUBlock_30090540876153
// MI455X (gfx1250) — matrix operand order repaired and checked by reading, not compiled, not run on hardware
//
#include <hip/hip_runtime.h>
#include <hip/hip_bf16.h>
#include <stdint.h>

// ---------------------------------------------------------------------------
// HSTU block, fused, for gfx1250 (MI455X). wave32, WMMA f32_16x16x32_f16,
// TDM (tensor_load_to_lds) staging for the projection GEMM B-tiles.
//   K0 convert_weights : uvqk -> uvqkT (f16, [2048][512]), W_o -> WoT (f16, [512][512])
//   K1 ln_proj         : LayerNorm(x) -> LDS f16 tile; B-tiles DMAd to LDS via
//                        TDM; WMMA GEMM; SiLU; scatter u(f32), v^T/q/k (f16)
//   K2 attn            : flash-style per-head attention with rel-pos +
//                        timestamp-bucket bias, SiLU/N, causal+pad masks
//   K3 ln_mul          : LayerNorm(attn) * u -> z (f16)
//   K4 out_gemm        : z @ W_o + b_o + x, pad-masked -> out (f32)
// ---------------------------------------------------------------------------

typedef __attribute__((ext_vector_type(16))) _Float16     v16h;
typedef __attribute__((ext_vector_type(8)))  _Float16     v8h;
typedef __attribute__((ext_vector_type(8)))  float        v8f;
typedef __attribute__((ext_vector_type(4)))  unsigned int u32x4;
typedef __attribute__((ext_vector_type(8)))  unsigned int u32x8;

#define B_  8
#define N_  1024
#define D_  512
#define H_  8
#define DV_ 64
#define DQ_ 64
#define E_  2048      // 2*H*DV + 2*H*DQ
#define NBKT_ 128
#define EPS_ 1e-5f

__device__ __forceinline__ float silu_f(float x) {
    // fast sigmoid: v_exp_f32 + v_rcp_f32 (avoid IEEE div expansion)
    return x * __builtin_amdgcn_rcpf(1.0f + __expf(-x));
}

__device__ __forceinline__ v8f wmma16(v16h a, v16h b, v8f c) {
    // D = A(16x32 f16) * B(32x16 f16) + C(16x16 f32)
    return __builtin_amdgcn_wmma_f32_16x16x32_f16(
        /*neg_a=*/false, a, /*neg_b=*/false, b,
        /*c_mod=*/(short)0, c, /*reuse_a=*/false, /*reuse_b=*/false);
}

__device__ __forceinline__ v16h pack16(v8h lo, v8h hi) {
    v16h a;
#pragma unroll
    for (int j = 0; j < 8; ++j) { a[j] = lo[j]; a[j + 8] = hi[j]; }
    return a;
}

// ---------------------------------------------------------------------------
// Tensor Data Mover: 2-D tile (16-bit elements) Global -> LDS.
// D# per CDNA5 ISA:
//   group0: count=1 | lds_addr | global_addr[56:0] | type=2
//   group1: wg_mask=0, data_size=1 (2B), dims/strides in element units
// 2-SGPR-group form (VADDR2/3 NULL): tensors up to 2D.
// A full DScnt drain precedes the issue: TDM LDS writes are unordered with
// the wave's own ds_loads, so pending reads of the staging buffer must have
// completed before the DMA may overwrite it (WAR).
// ---------------------------------------------------------------------------
__device__ __forceinline__ void tdm_load_tile_2d_f16(
        unsigned lds_byte_off, const _Float16* gptr,
        unsigned tensor_d0, unsigned tensor_d1,
        unsigned tile_d0, unsigned tile_d1, unsigned stride0_elems) {
    unsigned long long ga = (unsigned long long)(uintptr_t)gptr;
    u32x4 g0;
    g0[0] = 0x1u;                                              // count=1, user D#
    g0[1] = lds_byte_off;                                      // lds_addr
    g0[2] = (unsigned)ga;                                      // global_addr[31:0]
    g0[3] = ((unsigned)(ga >> 32) & 0x01FFFFFFu) | 0x80000000u; // addr[56:32]|type=2
    u32x8 g1;
    g1[0] = 0x00010000u;                                       // mask=0, data_size=2B
    g1[1] = (tensor_d0 & 0xFFFFu) << 16;                       // tensor_dim0 lo16
    g1[2] = (tensor_d0 >> 16) | ((tensor_d1 & 0xFFFFu) << 16); // d0 hi | d1 lo
    g1[3] = (tensor_d1 >> 16) | (tile_d0 << 16);               // d1 hi | tile_dim0
    g1[4] = tile_d1;                                           // tile_dim1, tile_dim2=0
    g1[5] = stride0_elems;                                     // dim0_stride lo32
    g1[6] = 0u;
    g1[7] = 0u;
    asm volatile("s_wait_dscnt 0x0\n\t"          // WAR: staging reads drained
                 "tensor_load_to_lds %0, %1"
                 :: "s"(g0), "s"(g1) : "memory");
}

// ---------------------------------------------------------------------------
// K0: weight conversion/transpose so WMMA B-fragments (lane=column, halves =
// contiguous K, K+16 for lanes 16..31) are contiguous 32B runs.
// ---------------------------------------------------------------------------
__global__ void convert_weights(const float* __restrict__ uvqk,
                                const float* __restrict__ W_o,
                                _Float16* __restrict__ uvqkT,
                                _Float16* __restrict__ WoT) {
    int idx = blockIdx.x * blockDim.x + threadIdx.x;
    int stride = gridDim.x * blockDim.x;
    for (int i = idx; i < E_ * D_; i += stride) {      // uvqkT[e][d] = uvqk[d][e]
        int e = i / D_, d = i - e * D_;
        uvqkT[i] = (_Float16)uvqk[(size_t)d * E_ + e];
    }
    for (int i = idx; i < D_ * D_; i += stride) {      // WoT[c][k] = W_o[k][c]
        int c = i / D_, k = i - c * D_;
        WoT[i] = (_Float16)W_o[(size_t)k * D_ + c];
    }
}

// ---------------------------------------------------------------------------
// K1: LayerNorm + uvqk projection + SiLU + scatter to u / vT / q / k.
// One block = 16 sequence rows; 8 waves; each wave owns 16 of the 128
// 16-column output tiles. B-tiles staged LDS<-global by TDM in 128-K chunks.
// ---------------------------------------------------------------------------
__device__ __forceinline__ void store_proj(float s, int g, int e,
                                           float* __restrict__ Ub,
                                           _Float16* __restrict__ Vt,
                                           _Float16* __restrict__ Qb,
                                           _Float16* __restrict__ Kb) {
    int b = g >> 10;          // g / N
    int n = g & (N_ - 1);
    if (e < 512) {
        Ub[(size_t)g * 512 + e] = s;
    } else if (e < 1024) {
        int ee = e - 512; int h = ee >> 6; int dv = ee & 63;
        Vt[((size_t)(b * H_ + h) * DV_ + dv) * N_ + n] = (_Float16)s;   // [B,H,DV,N]
    } else if (e < 1536) {
        int ee = e - 1024; int h = ee >> 6; int dq = ee & 63;
        Qb[((size_t)(b * H_ + h) * N_ + n) * DQ_ + dq] = (_Float16)s;   // [B,H,N,DQ]
    } else {
        int ee = e - 1536; int h = ee >> 6; int dq = ee & 63;
        Kb[((size_t)(b * H_ + h) * N_ + n) * DQ_ + dq] = (_Float16)s;
    }
}

__global__ void ln_proj(const float* __restrict__ x,
                        const _Float16* __restrict__ uvqkT,
                        const float* __restrict__ gamma_x,
                        const float* __restrict__ beta_x,
                        float* __restrict__ Ub, _Float16* __restrict__ Vt,
                        _Float16* __restrict__ Qb, _Float16* __restrict__ Kb) {
    __shared__ __align__(16) _Float16 sN[16][520];        // normed rows (16x512+pad)
    __shared__ __align__(16) _Float16 sB[8][16 * 128];    // per-wave TDM staging (4KB)

    int lane = threadIdx.x & 31;
    int wave = threadIdx.x >> 5;
    int tile0 = blockIdx.x * 16;                          // first global row

    // ---- Phase 1: LayerNorm, 2 rows per wave, wave32 shuffle reduction ----
    for (int rr = 0; rr < 2; ++rr) {
        int row = wave * 2 + rr;
        const float* xr = x + (size_t)(tile0 + row) * D_;
        float s = 0.f, s2 = 0.f;
        for (int i = lane; i < D_; i += 32) { float v = xr[i]; s += v; s2 += v * v; }
#pragma unroll
        for (int m = 16; m; m >>= 1) {
            s  += __shfl_xor(s,  m, 32);
            s2 += __shfl_xor(s2, m, 32);
        }
        float mu = s * (1.0f / D_);
        float var = s2 * (1.0f / D_) - mu * mu;
        float rstd = __builtin_amdgcn_rsqf(var + EPS_);
        for (int i = lane; i < D_; i += 32)
            sN[row][i] = (_Float16)((xr[i] - mu) * rstd * gamma_x[i] + beta_x[i]);
    }
    __syncthreads();

    // ---- Phase 2: WMMA GEMM: 16 rows x 2048 cols, K = 512 ----
    int arow = lane & 15;
    int koff = (lane >= 16) ? 16 : 0;   // A-layout hi-lane K offset
    int boff = (lane >= 16) ? 16 : 0;   // B-layout hi-lane K offset
    unsigned sb_off = (unsigned)(uintptr_t)&sB[wave][0];  // LDS byte offset

    for (int t = wave; t < E_ / 16; t += 8) {
        int e0 = t * 16;
        v8f c = {};
#pragma unroll
        for (int kc = 0; kc < 4; ++kc) {                 // 4 x (K=128) chunks
            // DMA 16 columns x 128 K-halves of uvqkT into this wave's buffer.
            tdm_load_tile_2d_f16(sb_off,
                                 uvqkT + (size_t)e0 * D_ + kc * 128,
                                 /*tensor_d0=*/D_, /*tensor_d1=*/E_,
                                 /*tile_d0=*/128, /*tile_d1=*/16,
                                 /*stride0=*/D_);
            __builtin_amdgcn_s_wait_tensorcnt(0);
#pragma unroll
            for (int kt2 = 0; kt2 < 4; ++kt2) {
                int kt = kc * 4 + kt2;
                v8h alo = *(const v8h*)&sN[arow][kt * 32 + koff];
                v8h ahi = *(const v8h*)&sN[arow][kt * 32 + 8 + koff];
                v16h a = pack16(alo, ahi);
                v16h b = *(const v16h*)&sB[wave][(lane & 15) * 128 + kt2 * 32 + boff];
                c = wmma16(a, b, c);
            }
        }
        int e = e0 + (lane & 15);
        int mrow = (lane >= 16) ? 8 : 0;
#pragma unroll
        for (int r = 0; r < 8; ++r)
            store_proj(silu_f(c[r]), tile0 + r + mrow, e, Ub, Vt, Qb, Kb);
    }
}

// ---------------------------------------------------------------------------
// K2: fused HSTU attention. One wave per (b, h, 16-row q tile).
// ---------------------------------------------------------------------------
__device__ __forceinline__ float hstu_score(float s, int b, int n, int m,
                                            const int* __restrict__ ts32,
                                            const unsigned char* __restrict__ pad,
                                            const float* __restrict__ ts_w,
                                            const float* __restrict__ pos_w) {
    if (m > n) return 0.0f;                       // causal: future masked
    if (pad[b * N_ + m]) return 0.0f;             // pad column masked
    int n1 = (n + 1 < N_) ? n + 1 : N_ - 1;       // ext[:, 1:] trick
    // timestamps < 1e7 -> low dword of the i64 is exact; diff fits i32 & f32
    int d = ts32[2 * (b * N_ + n1)] - ts32[2 * (b * N_ + m)];
    d = d < 0 ? -d : d;
    if (d < 1) d = 1;
    float logd = __logf((float)d) * (1.0f / 0.301f);
    int bkt = (int)logd;
    bkt = bkt < 0 ? 0 : (bkt > NBKT_ ? NBKT_ : bkt);
    float v = s + pos_w[(N_ - 1) + m - n] + ts_w[bkt];
    return silu_f(v) * (1.0f / (float)N_);
}

__global__ void attn_kernel(const _Float16* __restrict__ Qb,
                            const _Float16* __restrict__ Kb,
                            const _Float16* __restrict__ Vt,
                            const int* __restrict__ ts32,
                            const unsigned char* __restrict__ pad,
                            const float* __restrict__ ts_w,
                            const float* __restrict__ pos_w,
                            float* __restrict__ Ab) {
    __shared__ __align__(16) _Float16 sP[8][16][40];   // per-wave P staging

    int lane = threadIdx.x & 31;
    int wave = threadIdx.x >> 5;
    int gw = blockIdx.x * 8 + wave;        // 0 .. B*H*(N/16)-1 = 4095
    int qt = gw & 63;                      // N/16 = 64 tiles
    int h  = (gw >> 6) & (H_ - 1);
    int b  = gw >> 9;
    int n0 = qt * 16;

    const _Float16* qbase = Qb + (size_t)(b * H_ + h) * N_ * DQ_;
    const _Float16* kbase = Kb + (size_t)(b * H_ + h) * N_ * DQ_;
    const _Float16* vbase = Vt + (size_t)(b * H_ + h) * DV_ * N_;

    int arow = lane & 15;
    int koff = (lane >= 16) ? 16 : 0;
    int boff = (lane >= 16) ? 16 : 0;
    int mrow = (lane >= 16) ? 8 : 0;

    // Q A-fragments: dq 0..31 and 32..63
    v16h aq0, aq1;
    {
        const _Float16* qr = qbase + (size_t)(n0 + arow) * DQ_;
        aq0 = pack16(*(const v8h*)(qr + koff),      *(const v8h*)(qr + 8 + koff));
        aq1 = pack16(*(const v8h*)(qr + 32 + koff), *(const v8h*)(qr + 40 + koff));
    }

    v8f o0 = {}, o1 = {}, o2 = {}, o3 = {};

    for (int m0 = 0; m0 < n0 + 16; m0 += 32) {     // causal early-exit
        // ---- S = Q K^T, two 16x16 tiles (m0..m0+15, m0+16..m0+31) ----
        v8f s0 = {}, s1 = {};
        {
            const _Float16* kr0 = kbase + (size_t)(m0 + (lane & 15)) * DQ_;
            const _Float16* kr1 = kr0 + 16 * DQ_;
            s0 = wmma16(aq0, *(const v16h*)(kr0 + boff),      s0);
            s0 = wmma16(aq1, *(const v16h*)(kr0 + 32 + boff), s0);
            s1 = wmma16(aq0, *(const v16h*)(kr1 + boff),      s1);
            s1 = wmma16(aq1, *(const v16h*)(kr1 + 32 + boff), s1);
        }
        // ---- bias + SiLU + masks, stage P (f16, 16x32) in LDS ----
        int mcol = lane & 15;
#pragma unroll
        for (int r = 0; r < 8; ++r) {
            int n = n0 + r + mrow;
            sP[wave][r + mrow][mcol] =
                (_Float16)hstu_score(s0[r], b, n, m0 + mcol, ts32, pad, ts_w, pos_w);
            sP[wave][r + mrow][16 + mcol] =
                (_Float16)hstu_score(s1[r], b, n, m0 + 16 + mcol, ts32, pad, ts_w, pos_w);
        }
        // wave-private LDS round trip: wait on the CDNA5 DS counter before
        // re-reading the tile in WMMA A-fragment layout.
        asm volatile("s_wait_dscnt 0" ::: "memory");
        v16h ap = pack16(*(const v8h*)&sP[wave][arow][koff],
                         *(const v8h*)&sP[wave][arow][8 + koff]);
        // ---- O += P * V  (V^T rows are contiguous along m) ----
        const _Float16* vcol = vbase + (size_t)(lane & 15) * N_ + m0 + boff;
        o0 = wmma16(ap, *(const v16h*)(vcol),            o0);
        o1 = wmma16(ap, *(const v16h*)(vcol + 16 * N_),  o1);
        o2 = wmma16(ap, *(const v16h*)(vcol + 32 * N_),  o2);
        o3 = wmma16(ap, *(const v16h*)(vcol + 48 * N_),  o3);
    }

    // ---- write O (16 x 64) into attn [B,N,H*DV] ----
    float* outb = Ab + (size_t)b * N_ * (H_ * DV_) + h * DV_;
    int ncol = lane & 15;
#pragma unroll
    for (int r = 0; r < 8; ++r) {
        float* orow = outb + (size_t)(n0 + r + mrow) * (H_ * DV_);
        orow[ncol]      = o0[r];
        orow[16 + ncol] = o1[r];
        orow[32 + ncol] = o2[r];
        orow[48 + ncol] = o3[r];
    }
}

// ---------------------------------------------------------------------------
// K3: LayerNorm(attn) over H*DV, gated by u -> z (f16). One wave per row.
// ---------------------------------------------------------------------------
__global__ void ln_mul(const float* __restrict__ Ab, const float* __restrict__ Ub,
                       const float* __restrict__ gamma_a,
                       const float* __restrict__ beta_a,
                       _Float16* __restrict__ Zb) {
    int lane = threadIdx.x & 31;
    int wave = threadIdx.x >> 5;
    int row = blockIdx.x * 8 + wave;
    const float* ar = Ab + (size_t)row * D_;
    float s = 0.f, s2 = 0.f;
    for (int i = lane; i < D_; i += 32) { float v = ar[i]; s += v; s2 += v * v; }
#pragma unroll
    for (int m = 16; m; m >>= 1) {
        s  += __shfl_xor(s,  m, 32);
        s2 += __shfl_xor(s2, m, 32);
    }
    float mu = s * (1.0f / D_);
    float var = s2 * (1.0f / D_) - mu * mu;
    float rstd = __builtin_amdgcn_rsqf(var + EPS_);
    const float* ur = Ub + (size_t)row * D_;
    _Float16* zr = Zb + (size_t)row * D_;
    for (int i = lane; i < D_; i += 32) {
        float v = (ar[i] - mu) * rstd * gamma_a[i] + beta_a[i];
        zr[i] = (_Float16)(ur[i] * v);
    }
}

// ---------------------------------------------------------------------------
// K4: out = z @ W_o + b_o + x, pad rows zeroed. WMMA GEMM, A from global z.
// ---------------------------------------------------------------------------
__global__ void out_gemm(const _Float16* __restrict__ Zb,
                         const _Float16* __restrict__ WoT,
                         const float* __restrict__ b_o,
                         const float* __restrict__ x,
                         const unsigned char* __restrict__ pad,
                         float* __restrict__ out) {
    int lane = threadIdx.x & 31;
    int wave = threadIdx.x >> 5;
    int tile0 = blockIdx.x * 16;
    int arow = lane & 15;
    int koff = (lane >= 16) ? 16 : 0;
    int boff = (lane >= 16) ? 16 : 0;
    int mrow = (lane >= 16) ? 8 : 0;

    const _Float16* zr = Zb + (size_t)(tile0 + arow) * D_;
    for (int t = wave; t < D_ / 16; t += 8) {
        int e0 = t * 16;
        v8f c = {};
        const _Float16* bcol = WoT + (size_t)(e0 + (lane & 15)) * D_ + boff;
#pragma unroll 4
        for (int kt = 0; kt < 16; ++kt) {
            v16h a = pack16(*(const v8h*)(zr + kt * 32 + koff),
                            *(const v8h*)(zr + kt * 32 + 8 + koff));
            __builtin_prefetch(bcol + (kt + 1) * 32, 0, 1);   // global_prefetch_b8
            v16h b = *(const v16h*)(bcol + kt * 32);
            c = wmma16(a, b, c);
        }
        int e = e0 + (lane & 15);
#pragma unroll
        for (int r = 0; r < 8; ++r) {
            int g = tile0 + r + mrow;
            int b_idx = g >> 10;
            int n = g & (N_ - 1);
            float v = c[r] + b_o[e] + x[(size_t)g * D_ + e];
            out[(size_t)g * D_ + e] = pad[b_idx * N_ + n] ? 0.0f : v;
        }
    }
}

// ---------------------------------------------------------------------------
// Launcher. Inputs (setup_inputs order):
//  0 x f32[B,N,D]  1 timestamps i64[B,N]  2 causal_mask bool[N,N] (unused;
//  computed analytically)  3 pad_mask bool[B,N]  4 uvqk f32[512,2048]
//  5 W_o f32[512,512]  6 b_o f32[512]  7 gamma_x  8 beta_x  9 gamma_a
// 10 beta_a  11 ts_w f32[129]  12 pos_w f32[2047]
// ---------------------------------------------------------------------------
extern "C" void kernel_launch(void* const* d_in, const int* in_sizes, int n_in,
                              void* d_out, int out_size, void* d_ws, size_t ws_size,
                              hipStream_t stream) {
    const float*         x     = (const float*)d_in[0];
    const int*           ts32  = (const int*)d_in[1];    // low dwords of i64
    const unsigned char* pad   = (const unsigned char*)d_in[3];
    const float*         uvqk  = (const float*)d_in[4];
    const float*         W_o   = (const float*)d_in[5];
    const float*         b_o   = (const float*)d_in[6];
    const float*         g_x   = (const float*)d_in[7];
    const float*         be_x  = (const float*)d_in[8];
    const float*         g_a   = (const float*)d_in[9];
    const float*         be_a  = (const float*)d_in[10];
    const float*         ts_w  = (const float*)d_in[11];
    const float*         pos_w = (const float*)d_in[12];
    float*               out   = (float*)d_out;

    // Workspace layout (~66.5 MB total, all offsets 256B-aligned)
    char* w = (char*)d_ws;
    _Float16* uvqkT = (_Float16*)w;  w += (size_t)E_ * D_ * 2;            // 2 MB
    _Float16* WoT   = (_Float16*)w;  w += (size_t)D_ * D_ * 2;            // 0.5 MB
    _Float16* Qb    = (_Float16*)w;  w += (size_t)B_ * H_ * N_ * DQ_ * 2; // 8 MB
    _Float16* Kb    = (_Float16*)w;  w += (size_t)B_ * H_ * N_ * DQ_ * 2; // 8 MB
    _Float16* Vt    = (_Float16*)w;  w += (size_t)B_ * H_ * DV_ * N_ * 2; // 8 MB
    float*    Ub    = (float*)w;     w += (size_t)B_ * N_ * D_ * 4;       // 16 MB
    float*    Ab    = (float*)w;     w += (size_t)B_ * N_ * D_ * 4;       // 16 MB
    _Float16* Zb    = (_Float16*)w;  /* 8 MB */

    convert_weights<<<1024, 256, 0, stream>>>(uvqk, W_o, uvqkT, WoT);
    ln_proj<<<B_ * N_ / 16, 256, 0, stream>>>(x, uvqkT, g_x, be_x, Ub, Vt, Qb, Kb);
    attn_kernel<<<B_ * H_ * (N_ / 16) / 8, 256, 0, stream>>>(Qb, Kb, Vt, ts32, pad,
                                                             ts_w, pos_w, Ab);
    ln_mul<<<B_ * N_ / 8, 256, 0, stream>>>(Ab, Ub, g_a, be_a, Zb);
    out_gemm<<<B_ * N_ / 16, 256, 0, stream>>>(Zb, WoT, b_o, x, pad, out);
}
